// Model_7962869367240
// MI455X (gfx1250) — matrix operand order repaired and checked by reading, not compiled, not run on hardware
//
#include <hip/hip_runtime.h>

typedef __attribute__((ext_vector_type(16))) _Float16 v16h;
typedef __attribute__((ext_vector_type(8)))  float    v8f;

// ---------------------------------------------------------------------------
// WMMA GEMM:  C[M,N] = act(A[M,K] x B + bias) + Cadd        (f16 in, f32 acc)
//   A row-major (lda), BNT=1: B stored [N,K] (weights / idftT), BNT=0: B [K,N]
//   grid = (M/32, N/32, batch), block = 32: one wave owns a 32x32 tile
//   (2 A-frags x 2 B-frags -> 4 v_wmma per K-step; operand reuse 2x each).
//   Fragment layouts follow CDNA5 ISA 7.12.2 (16-bit A 16x32, B 32x16, f32 C).
// ---------------------------------------------------------------------------
template<int BNT, int ACT>
__global__ void gemm_wmma(const float* __restrict__ A, int lda, long long sA,
                          const float* __restrict__ Bm, int ldb, long long sB,
                          float* __restrict__ C, int ldc, long long sC,
                          int K,
                          const float* __restrict__ bias,
                          const float* __restrict__ Cadd, long long sCadd)
{
    const int lane = threadIdx.x & 31;
    const int tm = blockIdx.x, tn = blockIdx.y, bz = blockIdx.z;
    const float* Ab = A + (long long)bz * sA;
    const float* Bb = Bm + (long long)bz * sB;
    const int half = lane >> 4;
    const int l15  = lane & 15;
    const long long ar0 = (long long)tm * 32 + l15;
    const long long ar1 = ar0 + 16;
    const long long bc0 = (long long)tn * 32 + l15;
    const long long bc1 = bc0 + 16;
    v8f acc00 = {}, acc01 = {}, acc10 = {}, acc11 = {};
    for (int k0 = 0; k0 < K; k0 += 32) {
        v16h a0, a1, b0, b1;
        const int ka = k0 + half * 8;
        #pragma unroll
        for (int e = 0; e < 16; ++e) {
            int k = ka + e + ((e & 8) ? 8 : 0);   // A: lane row, K {0..7,16..23}(+8)
            a0[e] = (_Float16)Ab[ar0 * lda + k];
            a1[e] = (_Float16)Ab[ar1 * lda + k];
        }
        const int kb = k0 + half * 8;
        #pragma unroll
        for (int e = 0; e < 16; ++e) {
            int k = kb + e + ((e & 8) ? 8 : 0);   // B: lane col, K {0..7,16..23}(+8)
            float v0 = BNT ? Bb[bc0 * (long long)ldb + k]
                           : Bb[(long long)k * ldb + bc0];
            float v1 = BNT ? Bb[bc1 * (long long)ldb + k]
                           : Bb[(long long)k * ldb + bc1];
            b0[e] = (_Float16)v0;
            b1[e] = (_Float16)v1;
        }
        if (k0 + 32 < K) {                         // gfx1250 global_prefetch path
            __builtin_prefetch(&Ab[ar0 * lda + k0 + 32], 0, 1);
            __builtin_prefetch(&Ab[ar1 * lda + k0 + 32], 0, 1);
            if (BNT) {
                __builtin_prefetch(&Bb[bc0 * (long long)ldb + k0 + 32], 0, 1);
                __builtin_prefetch(&Bb[bc1 * (long long)ldb + k0 + 32], 0, 1);
            } else {
                __builtin_prefetch(&Bb[(long long)(k0 + 32) * ldb + bc0], 0, 1);
            }
        }
        acc00 = __builtin_amdgcn_wmma_f32_16x16x32_f16(false, a0, false, b0,
                                                       (short)0, acc00, false, false);
        acc01 = __builtin_amdgcn_wmma_f32_16x16x32_f16(false, a0, false, b1,
                                                       (short)0, acc01, false, false);
        acc10 = __builtin_amdgcn_wmma_f32_16x16x32_f16(false, a1, false, b0,
                                                       (short)0, acc10, false, false);
        acc11 = __builtin_amdgcn_wmma_f32_16x16x32_f16(false, a1, false, b1,
                                                       (short)0, acc11, false, false);
    }
    const int crow = tm * 32 + half * 8;
    const int ccol = tn * 32 + l15;
    float* Cb = C + (long long)bz * sC;
    const float* Ra = Cadd ? (Cadd + (long long)bz * sCadd) : nullptr;
    #pragma unroll
    for (int mi = 0; mi < 2; ++mi) {
        #pragma unroll
        for (int ni = 0; ni < 2; ++ni) {
            v8f acc = (mi == 0) ? (ni == 0 ? acc00 : acc01)
                                : (ni == 0 ? acc10 : acc11);
            const int cc = ccol + ni * 16;
            const float bv = bias ? bias[cc] : 0.f;
            #pragma unroll
            for (int r = 0; r < 8; ++r) {
                const long long rr = crow + mi * 16 + r;
                float v = acc[r] + bv;
                if (ACT == 1) v = 0.5f * v * (1.f + erff(v * 0.7071067811865475f));
                if (Ra) v += Ra[rr * ldc + cc];
                Cb[rr * ldc + cc] = v;
            }
        }
    }
}

// ------------------------- DFT tables (exact phases) ------------------------
__global__ void init_tables_kernel(float* __restrict__ cosT, float* __restrict__ sinT,
                                   float* __restrict__ idftT)
{
    long long i = (long long)blockIdx.x * blockDim.x + threadIdx.x;
    if (i >= 64LL * 4096) return;
    int l = (int)(i & 4095);
    int m = (int)(i >> 12);
    int p = (m * l) & 4095;                       // exact integer phase mod L
    float th = (float)p * (6.2831853071795864769f / 4096.f);
    float cv = cosf(th), sv = sinf(th);
    cosT[i] = cv;                                 // Xr = sum x*cos
    sinT[i] = -sv;                                // Xi = sum x*(-sin)  (rfft)
    float cm = (m == 0 ? 1.f : 2.f) / 4096.f;     // irfft with modes < L/2 only
    idftT[(long long)l * 128 + m]      = cm * cv;
    idftT[(long long)l * 128 + 64 + m] = -cm * sv;
}

// ----------------- circular conv, kernel size 3 (torch layout) --------------
template<int ACC>
__global__ void conv3_kernel(const float* __restrict__ x, const float* __restrict__ w,
                             float* __restrict__ out, int Lx, int Cin, int Cout,
                             long long total)
{
    long long i = (long long)blockIdx.x * blockDim.x + threadIdx.x;
    if (i >= total) return;
    int co = (int)(i % Cout);
    long long bl = i / Cout;
    int l = (int)(bl % Lx);
    int b = (int)(bl / Lx);
    const float* xb = x + (long long)b * Lx * Cin;
    float acc = 0.f;
    for (int t = 0; t < 3; ++t) {
        int lp = l + t - 1;
        lp = (lp < 0) ? lp + Lx : (lp >= Lx ? lp - Lx : lp);
        const float* xr = xb + (long long)lp * Cin;
        const float* wr = w + ((long long)co * Cin) * 3 + t;
        for (int ci = 0; ci < Cin; ++ci) acc += xr[ci] * wr[(long long)ci * 3];
    }
    if (ACC) out[i] += acc; else out[i] = acc;
}

// ---------------- series_decomp: res = x - movavg25(x), m optional ----------
__global__ void decomp_kernel(const float* __restrict__ x, float* __restrict__ res,
                              float* __restrict__ mout, int Lx, int C, long long total)
{
    long long i = (long long)blockIdx.x * blockDim.x + threadIdx.x;
    if (i >= total) return;
    int c = (int)(i % C);
    long long bl = i / C;
    int l = (int)(bl % Lx);
    int b = (int)(bl / Lx);
    const float* xb = x + ((long long)b * Lx) * C + c;
    float s = 0.f;
    for (int j = -12; j <= 12; ++j) {
        int lj = l + j;
        lj = lj < 0 ? 0 : (lj >= Lx ? Lx - 1 : lj);
        s += xb[(long long)lj * C];
    }
    float m = s * (1.f / 25.f);
    res[i] = x[i] - m;
    if (mout) mout[i] = m;
}

// -------- complex mode mixing: out[b,h,o,m] = sum_i X[b,h,i,m] * w[h,i,o,m] -
// X given as Xr/Xi[b][m][h*64+i]; writes Aif[b][h*64+o][{m, 64+m}] * scale
__global__ void fourier_mul_kernel(const float* __restrict__ Xr, const float* __restrict__ Xi,
                                   const float* __restrict__ wr, const float* __restrict__ wi,
                                   float* __restrict__ aif, float scale)
{
    long long i = (long long)blockIdx.x * blockDim.x + threadIdx.x;
    if (i >= 16LL * 8 * 64 * 64) return;
    int m = (int)(i & 63), o = (int)((i >> 6) & 63), h = (int)((i >> 12) & 7), b = (int)(i >> 15);
    const float* xr = Xr + (((long long)b * 64 + m) * 512) + h * 64;
    const float* xi = Xi + (((long long)b * 64 + m) * 512) + h * 64;
    const float* wrp = wr + (((long long)h * 64) * 64 + o) * 64 + m;
    const float* wip = wi + (((long long)h * 64) * 64 + o) * 64 + m;
    float ar = 0.f, ai = 0.f;
    for (int e = 0; e < 64; ++e) {
        float xrr = xr[e], xii = xi[e];
        float wrr = wrp[(long long)e * 4096], wii = wip[(long long)e * 4096];
        ar += xrr * wrr - xii * wii;
        ai += xrr * wii + xii * wrr;
    }
    float* dst = aif + (((long long)b * 512) + h * 64 + o) * 128;
    dst[m]      = ar * scale;
    dst[64 + m] = ai * scale;
}

// ------- xqk[b,h,x,y] = ctanh( sum_e Xq[b,h,e,x]*Xk[b,h,e,y] )  (complex) ---
__global__ void xqk_kernel(const float* __restrict__ Xqr, const float* __restrict__ Xqi,
                           const float* __restrict__ Xkr, const float* __restrict__ Xki,
                           float* __restrict__ QKr, float* __restrict__ QKi)
{
    long long i = (long long)blockIdx.x * blockDim.x + threadIdx.x;
    if (i >= 16LL * 8 * 64 * 64) return;
    int y = (int)(i & 63), x = (int)((i >> 6) & 63), h = (int)((i >> 12) & 7), b = (int)(i >> 15);
    const float* qr = Xqr + ((long long)b * 64 + x) * 512 + h * 64;
    const float* qi = Xqi + ((long long)b * 64 + x) * 512 + h * 64;
    const float* kr = Xkr + ((long long)b * 64 + y) * 512 + h * 64;
    const float* ki = Xki + ((long long)b * 64 + y) * 512 + h * 64;
    float ar = 0.f, ai = 0.f;
    for (int e = 0; e < 64; ++e) {
        float a = qr[e], bb = qi[e], c = kr[e], d = ki[e];
        ar += a * c - bb * d;
        ai += a * d + bb * c;
    }
    // stable complex tanh
    float twoa = 2.f * ar, twob = 2.f * ai, tr, ti;
    if (fabsf(twoa) > 18.f) { tr = twoa > 0.f ? 1.f : -1.f; ti = 0.f; }
    else {
        float den = coshf(twoa) + cosf(twob);
        tr = sinhf(twoa) / den;
        ti = sinf(twob) / den;
    }
    QKr[i] = tr; QKi[i] = ti;
}

// -- xqkv[b,h,e,x] = sum_y QK[b,h,x,y]*Xk[b,h,e,y]; store V[b][x][h*64+e] ----
__global__ void xqkv_kernel(const float* __restrict__ QKr, const float* __restrict__ QKi,
                            const float* __restrict__ Xkr, const float* __restrict__ Xki,
                            float* __restrict__ Vr, float* __restrict__ Vi)
{
    long long i = (long long)blockIdx.x * blockDim.x + threadIdx.x;
    if (i >= 16LL * 8 * 64 * 64) return;
    int x = (int)(i & 63), e = (int)((i >> 6) & 63), h = (int)((i >> 12) & 7), b = (int)(i >> 15);
    const float* qkr = QKr + (((long long)b * 8 + h) * 64 + x) * 64;
    const float* qki = QKi + (((long long)b * 8 + h) * 64 + x) * 64;
    const float* kr = Xkr + (long long)b * 64 * 512 + h * 64 + e;
    const float* ki = Xki + (long long)b * 64 * 512 + h * 64 + e;
    float ar = 0.f, ai = 0.f;
    for (int y = 0; y < 64; ++y) {
        float a = qkr[y], bb = qki[y];
        float c = kr[(long long)y * 512], d = ki[(long long)y * 512];
        ar += a * c - bb * d;
        ai += a * d + bb * c;
    }
    long long o = ((long long)b * 64 + x) * 512 + h * 64 + e;
    Vr[o] = ar; Vi[o] = ai;
}

// ---------------- layernorm pieces (my_layernorm) ---------------------------
__global__ void ln_rows_kernel(const float* __restrict__ x, const float* __restrict__ g,
                               const float* __restrict__ bb, float* __restrict__ out)
{
    long long row = blockIdx.x;
    int t = threadIdx.x;                          // 256 threads, C = 512
    const float* xr = x + row * 512;
    __shared__ float sm[256];
    __shared__ float sq[256];
    float v0 = xr[t], v1 = xr[t + 256];
    sm[t] = v0 + v1;
    sq[t] = v0 * v0 + v1 * v1;
    __syncthreads();
    for (int s = 128; s > 0; s >>= 1) {
        if (t < s) { sm[t] += sm[t + s]; sq[t] += sq[t + s]; }
        __syncthreads();
    }
    float mu = sm[0] * (1.f / 512.f);
    float var = sq[0] * (1.f / 512.f) - mu * mu;
    float rs = rsqrtf(var + 1e-5f);
    out[row * 512 + t]       = (v0 - mu) * rs * g[t] + bb[t];
    out[row * 512 + t + 256] = (v1 - mu) * rs * g[t + 256] + bb[t + 256];
}

__global__ void col_mean_kernel(const float* __restrict__ x, float* __restrict__ mean,
                                int Lx, int C, int n)
{
    int i = blockIdx.x * blockDim.x + threadIdx.x;
    if (i >= n) return;
    int c = i % C, b = i / C;
    const float* xb = x + (long long)b * Lx * C + c;
    float s = 0.f;
    for (int l = 0; l < Lx; ++l) s += xb[(long long)l * C];
    mean[i] = s / (float)Lx;
}

__global__ void sub_col_kernel(float* __restrict__ x, const float* __restrict__ mean,
                               int Lx, int C, long long total)
{
    long long i = (long long)blockIdx.x * blockDim.x + threadIdx.x;
    if (i >= total) return;
    int c = (int)(i % C);
    int b = (int)((i / C) / Lx);
    x[i] -= mean[b * C + c];
}

// --------------- trend init: movavg(x_enc) tail + mean repeat ---------------
__global__ void trend_init_kernel(const float* __restrict__ xenc, const float* __restrict__ xem,
                                  float* __restrict__ trend)
{
    long long i = (long long)blockIdx.x * blockDim.x + threadIdx.x;
    if (i >= 16LL * 4096 * 21) return;
    int c = (int)(i % 21);
    long long bl = i / 21;
    int l = (int)(bl % 4096);
    int b = (int)(bl / 4096);
    float v;
    if (l < 2048) {
        int lc = 2048 + l;                        // last LABEL_LEN of trend_init
        const float* xb = xenc + ((long long)b * 4096) * 21 + c;
        float s = 0.f;
        for (int j = -12; j <= 12; ++j) {
            int lj = lc + j;
            lj = lj < 0 ? 0 : (lj > 4095 ? 4095 : lj);
            s += xb[(long long)lj * 21];
        }
        v = s * (1.f / 25.f);
    } else {
        v = xem[b * 21 + c];
    }
    trend[i] = v;
}

// ---------------- final: out = trend + dec @ proj_w.T + proj_b, tail --------
__global__ void final_kernel(const float* __restrict__ dec, const float* __restrict__ trend,
                             const float* __restrict__ pw, const float* __restrict__ pb,
                             float* __restrict__ out)
{
    long long i = (long long)blockIdx.x * blockDim.x + threadIdx.x;
    if (i >= 16LL * 2048 * 21) return;
    int c = (int)(i % 21);
    long long bl = i / 21;
    int l2 = (int)(bl % 2048);
    int b = (int)(bl / 2048);
    int l = 2048 + l2;
    const float* xr = dec + ((long long)b * 4096 + l) * 512;
    const float* wr = pw + (long long)c * 512;
    float s = pb[c];
    for (int k = 0; k < 512; ++k) s += xr[k] * wr[k];
    out[i] = s + trend[((long long)b * 4096 + l) * 21 + c];
}

// ===========================================================================
extern "C" void kernel_launch(void* const* d_in, const int* in_sizes, int n_in,
                              void* d_out, int out_size, void* d_ws, size_t ws_size,
                              hipStream_t stream)
{
    (void)n_in; (void)out_size;
    // -------- parameter map (detect flattening convention) ------------------
    const float *x_enc, *x_dec, *enc_emb, *dec_emb;
    struct EncL { const float *wq,*bq,*wo,*bo,*fwr,*fwi,*w1,*w2; } el[2];
    struct DecL { const float *swq,*sbq,*swo,*sbo,*sfwr,*sfwi;
                  const float *cwq,*cbq,*cwo,*cbo,*cfwr,*cfwi,*cwk,*cbk;
                  const float *w1,*w2,*tproj; } dl;
    const float *eng,*enb,*dng,*dnb,*pw,*pb;
    auto P = [&](int i){ return (const float*)d_in[i]; };
    if (in_sizes[0] == 16 * 4096 * 21) {
        // dict insertion order: x_enc, x_dec, then params in construction order
        x_enc = P(0); x_dec = P(1); enc_emb = P(2); dec_emb = P(3);
        for (int i = 0; i < 2; ++i) {
            int o = 4 + i * 8;
            el[i] = { P(o), P(o+1), P(o+2), P(o+3), P(o+4), P(o+5), P(o+6), P(o+7) };
        }
        dl.swq = P(20); dl.sbq = P(21); dl.swo = P(22); dl.sbo = P(23);
        dl.sfwr = P(24); dl.sfwi = P(25);
        dl.cwq = P(26); dl.cbq = P(27); dl.cwo = P(28); dl.cbo = P(29);
        dl.cfwr = P(30); dl.cfwi = P(31); dl.cwk = P(32); dl.cbk = P(33);
        dl.w1 = P(34); dl.w2 = P(35); dl.tproj = P(36);
        eng = P(37); enb = P(38); dng = P(39); dnb = P(40); pw = P(41); pb = P(42);
    } else {
        // jax tree (sorted keys) order: params(sorted), x_dec, x_enc
        dec_emb = P(0);
        dl.cbk = P(1); dl.cbo = P(2); dl.cbq = P(3); dl.cfwi = P(4); dl.cfwr = P(5);
        dl.cwk = P(6); dl.cwo = P(7); dl.cwq = P(8);
        dl.sbo = P(9); dl.sbq = P(10); dl.sfwi = P(11); dl.sfwr = P(12);
        dl.swo = P(13); dl.swq = P(14);
        dl.tproj = P(15); dl.w1 = P(16); dl.w2 = P(17);
        dnb = P(18); dng = P(19); enc_emb = P(20);
        for (int i = 0; i < 2; ++i) {
            int o = 21 + i * 8;
            el[i].bo = P(o); el[i].bq = P(o+1); el[i].fwi = P(o+2); el[i].fwr = P(o+3);
            el[i].w1 = P(o+4); el[i].w2 = P(o+5); el[i].wo = P(o+6); el[i].wq = P(o+7);
        }
        enb = P(37); eng = P(38); pb = P(39); pw = P(40); x_dec = P(41); x_enc = P(42);
    }

    // -------- workspace layout (floats) -------------------------------------
    const long long NACT = 16LL * 4096 * 512;
    float* ws = (float*)d_ws;
    long long off = 0;
    float* COSp  = ws + off; off += 64LL * 4096;
    float* SINp  = ws + off; off += 64LL * 4096;
    float* IDFTp = ws + off; off += 4096LL * 128;
    float* XRp   = ws + off; off += 16LL * 64 * 512;
    float* XIp   = ws + off; off += 16LL * 64 * 512;
    float* XR2p  = ws + off; off += 16LL * 64 * 512;
    float* XI2p  = ws + off; off += 16LL * 64 * 512;
    float* AIFp  = ws + off; off += 16LL * 512 * 128;
    float* QKRp  = ws + off; off += 16LL * 8 * 64 * 64;
    float* QKIp  = ws + off; off += 16LL * 8 * 64 * 64;
    float* VRp   = ws + off; off += 16LL * 64 * 512;
    float* VIp   = ws + off; off += 16LL * 64 * 512;
    float* TRNDp = ws + off; off += 16LL * 4096 * 21;
    float* CMNp  = ws + off; off += 16LL * 512;
    float* XEMp  = ws + off; off += 16LL * 21;
    float* H1p   = ws + off; off += 8192LL * 2048;
    float* X0p   = ws + off; off += NACT;
    float* S1p   = ws + off; off += NACT;
    float* S2p   = ws + off; off += NACT;
    float* Yp    = ws + off; off += NACT;
    float* EOUTp = ws + off; off += NACT;
    if ((size_t)off * sizeof(float) > ws_size) return;   // deterministic guard

    const int TPB = 256;
    auto blocks = [](long long n){ return (unsigned)((n + 255) / 256); };

    // -------- DFT / iDFT tables ---------------------------------------------
    init_tables_kernel<<<blocks(64LL * 4096), TPB, 0, stream>>>(COSp, SINp, IDFTp);

    // -------- helper sequences ----------------------------------------------
    auto gemm_proj = [&](const float* A, const float* W, const float* bias,
                         const float* Cadd, float* C, hipStream_t st) {
        gemm_wmma<1,0><<<dim3(2048, 16, 1), 32, 0, st>>>(           // M=65536,N=512
            A, 512, 0, W, 512, 0, C, 512, 0, 512, bias, Cadd, 0);
    };
    auto dft_fwd = [&](const float* q, float* Xr, float* Xi) {      // M=64,N=512,K=4096
        gemm_wmma<0,0><<<dim3(2, 16, 16), 32, 0, stream>>>(
            COSp, 4096, 0, q, 512, 4096LL * 512, Xr, 512, 64LL * 512, 4096,
            nullptr, nullptr, 0);
        gemm_wmma<0,0><<<dim3(2, 16, 16), 32, 0, stream>>>(
            SINp, 4096, 0, q, 512, 4096LL * 512, Xi, 512, 64LL * 512, 4096,
            nullptr, nullptr, 0);
    };
    auto dft_inv = [&](float* outBHEL) {     // AIFp -> [B][512][4096] (== [B][L][512] view)
        gemm_wmma<1,0><<<dim3(16, 128, 16), 32, 0, stream>>>(       // M=512,N=4096,K=128
            AIFp, 128, 512LL * 128, IDFTp, 128, 0, outBHEL, 4096, 512LL * 4096, 128,
            nullptr, nullptr, 0);
    };
    auto fourier_self = [&](const float* Xin, const float* wq, const float* bq,
                            const float* wo, const float* bo,
                            const float* fwr, const float* fwi, float* Yout) {
        gemm_proj(Xin, wq, bq, nullptr, S1p, stream);              // q
        dft_fwd(S1p, XRp, XIp);
        fourier_mul_kernel<<<blocks(16LL*8*64*64), TPB, 0, stream>>>(
            XRp, XIp, fwr, fwi, AIFp, 1.f);
        dft_inv(S1p);                                              // attn out (reshaped view)
        gemm_proj(S1p, wo, bo, Xin, Yout, stream);                 // x + out@wo.T + bo
    };
    auto ffn = [&](const float* Xin, const float* w1, const float* w2, float* Yout) {
        for (int c = 0; c < 8; ++c) {                              // 8192-row chunks
            long long o2 = (long long)c * 8192 * 512;
            gemm_wmma<1,1><<<dim3(256, 64, 1), 32, 0, stream>>>(   // M=8192,N=2048
                Xin + o2, 512, 0, w1, 512, 0, H1p, 2048, 0, 512, nullptr, nullptr, 0);
            gemm_wmma<1,0><<<dim3(256, 16, 1), 32, 0, stream>>>(   // M=8192,N=512
                H1p, 2048, 0, w2, 2048, 0, Yp + o2, 512, 0, 2048, nullptr, Xin + o2, 0);
        }
        (void)Yout;
    };

    // ============================ ENCODER ====================================
    conv3_kernel<0><<<blocks(NACT), TPB, 0, stream>>>(x_enc, enc_emb, X0p, 4096, 21, 512, NACT);
    for (int li = 0; li < 2; ++li) {
        fourier_self(X0p, el[li].wq, el[li].bq, el[li].wo, el[li].bo,
                     el[li].fwr, el[li].fwi, Yp);
        decomp_kernel<<<blocks(NACT), TPB, 0, stream>>>(Yp, S2p, nullptr, 4096, 512, NACT);
        ffn(S2p, el[li].w1, el[li].w2, Yp);                        // Yp = S2 + ffn(S2)
        decomp_kernel<<<blocks(NACT), TPB, 0, stream>>>(Yp, X0p, nullptr, 4096, 512, NACT);
    }
    ln_rows_kernel<<<65536, 256, 0, stream>>>(X0p, eng, enb, EOUTp);
    col_mean_kernel<<<blocks(16 * 512), TPB, 0, stream>>>(EOUTp, CMNp, 4096, 512, 16 * 512);
    sub_col_kernel<<<blocks(NACT), TPB, 0, stream>>>(EOUTp, CMNp, 4096, 512, NACT);

    // ============================ DECODER ====================================
    conv3_kernel<0><<<blocks(NACT), TPB, 0, stream>>>(x_dec, dec_emb, X0p, 4096, 21, 512, NACT);
    col_mean_kernel<<<blocks(16 * 21), TPB, 0, stream>>>(x_enc, XEMp, 4096, 21, 16 * 21);
    trend_init_kernel<<<blocks(16LL * 4096 * 21), TPB, 0, stream>>>(x_enc, XEMp, TRNDp);

    // --- self attention + decomp (t1) ---
    fourier_self(X0p, dl.swq, dl.sbq, dl.swo, dl.sbo, dl.sfwr, dl.sfwi, Yp);
    decomp_kernel<<<blocks(NACT), TPB, 0, stream>>>(Yp, S2p, S1p, 4096, 512, NACT);
    conv3_kernel<1><<<blocks(16LL * 4096 * 21), TPB, 0, stream>>>(
        S1p, dl.tproj, TRNDp, 4096, 512, 21, 16LL * 4096 * 21);

    // --- cross attention ---
    gemm_proj(S2p, dl.cwq, dl.cbq, nullptr, Yp, stream);           // q
    dft_fwd(Yp, XRp, XIp);
    gemm_proj(EOUTp, dl.cwk, dl.cbk, nullptr, Yp, stream);         // k from encoder out
    dft_fwd(Yp, XR2p, XI2p);
    xqk_kernel<<<blocks(16LL*8*64*64), TPB, 0, stream>>>(XRp, XIp, XR2p, XI2p, QKRp, QKIp);
    xqkv_kernel<<<blocks(16LL*8*64*64), TPB, 0, stream>>>(QKRp, QKIp, XR2p, XI2p, VRp, VIp);
    fourier_mul_kernel<<<blocks(16LL*8*64*64), TPB, 0, stream>>>(
        VRp, VIp, dl.cfwr, dl.cfwi, AIFp, 1.f / (512.f * 512.f));
    dft_inv(S1p);
    gemm_proj(S1p, dl.cwo, dl.cbo, S2p, Yp, stream);               // x + cross@wo.T + bo
    decomp_kernel<<<blocks(NACT), TPB, 0, stream>>>(Yp, X0p, S1p, 4096, 512, NACT);   // t2
    conv3_kernel<1><<<blocks(16LL * 4096 * 21), TPB, 0, stream>>>(
        S1p, dl.tproj, TRNDp, 4096, 512, 21, 16LL * 4096 * 21);

    // --- FFN + decomp (t3) ---
    ffn(X0p, dl.w1, dl.w2, Yp);                                    // Yp = X0 + ffn(X0)
    decomp_kernel<<<blocks(NACT), TPB, 0, stream>>>(Yp, S2p, S1p, 4096, 512, NACT);
    conv3_kernel<1><<<blocks(16LL * 4096 * 21), TPB, 0, stream>>>(
        S1p, dl.tproj, TRNDp, 4096, 512, 21, 16LL * 4096 * 21);

    // --- decoder norm, projection, trend add, tail slice ---
    ln_rows_kernel<<<65536, 256, 0, stream>>>(S2p, dng, dnb, Yp);
    col_mean_kernel<<<blocks(16 * 512), TPB, 0, stream>>>(Yp, CMNp, 4096, 512, 16 * 512);
    sub_col_kernel<<<blocks(NACT), TPB, 0, stream>>>(Yp, CMNp, 4096, 512, NACT);
    final_kernel<<<blocks(16LL * 2048 * 21), TPB, 0, stream>>>(
        Yp, TRNDp, pw, pb, (float*)d_out);
}
